// MultiHeadAttention_61675730370610
// MI455X (gfx1250) — matrix operand order repaired and checked by reading, not compiled, not run on hardware
//
#include <hip/hip_runtime.h>

// ---------------------------------------------------------------------------
// Types for CDNA5 WMMA (wave32): v_wmma_f32_16x16x32_bf16
// ---------------------------------------------------------------------------
typedef __bf16 bf16x16 __attribute__((ext_vector_type(16)));
typedef __bf16 bf16x8  __attribute__((ext_vector_type(8)));
typedef float  f32x8   __attribute__((ext_vector_type(8)));

#define B_  2
#define S_  2048
#define D_  1024
#define H_  16
#define HD_ 64

static __device__ inline __bf16 f2bf(float f) {
    union { float f; unsigned u; } v; v.f = f;
    unsigned r = v.u + 0x7FFFu + ((v.u >> 16) & 1u);   // round-to-nearest-even
    unsigned short h = (unsigned short)(r >> 16);
    return __builtin_bit_cast(__bf16, h);
}

// A-fragment (16x32 bf16, M x K): lane<16 -> row=lane, K {0..15};
// lane>=16 -> row=lane-16, K {16..31}. Caller pre-offsets pointer by
// (lane&15)*ld + (lane>>4)*16; we load [0..7] and [8..15] relative to that.
static __device__ inline bf16x16 load_afrag(const __bf16* p) {
    bf16x8 lo = *(const bf16x8*)(p);
    bf16x8 hi = *(const bf16x8*)(p + 8);
    bf16x16 r;
#pragma unroll
    for (int i = 0; i < 8; ++i) { r[i] = lo[i]; r[8 + i] = hi[i]; }
    return r;
}

// B-fragment (32x16 bf16, K x N): lane holds fixed N=lane&15 and 16 contiguous
// K values starting at (lane>>4)*16. Caller passes fully-offset pointer.
static __device__ inline bf16x16 load_bfrag(const __bf16* p) {
    return *(const bf16x16*)p;   // 32B contiguous -> 2x b128
}

static __device__ inline f32x8 wmma_bf16(bf16x16 a, bf16x16 b, f32x8 c) {
    return __builtin_amdgcn_wmma_f32_16x16x32_bf16(
        false, a, false, b, (short)0, c, false, false);
}

// ---------------------------------------------------------------------------
// 1) LayerNorm -> bf16   (one block of 256 per row, D=1024 -> float4/thread)
// ---------------------------------------------------------------------------
__global__ void ln_kernel(const float* __restrict__ x,
                          const float* __restrict__ gamma,
                          const float* __restrict__ beta,
                          __bf16* __restrict__ xn) {
    int row = blockIdx.x;
    const float4* xr = (const float4*)(x + (size_t)row * D_);
    float4 v = xr[threadIdx.x];
    float s  = v.x + v.y + v.z + v.w;
    float ss = v.x * v.x + v.y * v.y + v.z * v.z + v.w * v.w;
#pragma unroll
    for (int off = 16; off; off >>= 1) {
        s  += __shfl_xor(s,  off, 32);
        ss += __shfl_xor(ss, off, 32);
    }
    __shared__ float rs[8], rss[8];
    int wid = threadIdx.x >> 5;
    if ((threadIdx.x & 31) == 0) { rs[wid] = s; rss[wid] = ss; }
    __syncthreads();
    s = 0.f; ss = 0.f;
#pragma unroll
    for (int w = 0; w < 8; ++w) { s += rs[w]; ss += rss[w]; }
    float mean = s * (1.0f / D_);
    float var  = ss * (1.0f / D_) - mean * mean;
    float inv  = rsqrtf(var + 1e-5f);
    float4 g  = ((const float4*)gamma)[threadIdx.x];
    float4 bt = ((const float4*)beta)[threadIdx.x];
    __bf16* o = xn + (size_t)row * D_ + threadIdx.x * 4;
    o[0] = f2bf((v.x - mean) * inv * g.x + bt.x);
    o[1] = f2bf((v.y - mean) * inv * g.y + bt.y);
    o[2] = f2bf((v.z - mean) * inv * g.z + bt.z);
    o[3] = f2bf((v.w - mean) * inv * g.w + bt.w);
}

// ---------------------------------------------------------------------------
// 2) Weight transpose + f32->bf16:  Wt[n*D+k] = W[k*D+n]   (tiled via LDS)
//    block (32,8), grid (32,32)
// ---------------------------------------------------------------------------
__global__ void wtrans(const float* __restrict__ W, __bf16* __restrict__ Wt) {
    __shared__ float tile[32][33];
    int kb = blockIdx.y * 32, nb = blockIdx.x * 32;
    int tx = threadIdx.x, ty = threadIdx.y;
#pragma unroll
    for (int j = 0; j < 4; ++j)
        tile[ty + 8 * j][tx] = W[(size_t)(kb + ty + 8 * j) * D_ + nb + tx];
    __syncthreads();
#pragma unroll
    for (int j = 0; j < 4; ++j)
        Wt[(size_t)(nb + ty + 8 * j) * D_ + kb + tx] = f2bf(tile[tx][ty + 8 * j]);
}

// ---------------------------------------------------------------------------
// 3) bf16 GEMM: C[4096,1024] = A[4096,1024] @ Wt^T + bias -> bf16
//    wave computes 32x64 (2 A frags reuse each B frag); block = 4 waves
//    covering 256 N-cols; grid (128, 4)
// ---------------------------------------------------------------------------
__global__ void gemm_qkv(const __bf16* __restrict__ A,
                         const __bf16* __restrict__ Wt,
                         const float* __restrict__ bias,
                         __bf16* __restrict__ C) {
    int m0   = blockIdx.x * 32;
    int wave = threadIdx.x >> 5;
    int lane = threadIdx.x & 31;
    int lm = lane & 15, hf = lane >> 4;
    int n0 = blockIdx.y * 256 + wave * 64;

    const __bf16* arow0 = A + (size_t)(m0 + lm) * D_ + hf * 16;
    const __bf16* arow1 = arow0 + (size_t)16 * D_;
    f32x8 acc[2][4] = {};
    for (int k = 0; k < D_; k += 32) {
        bf16x16 a0 = load_afrag(arow0 + k);
        bf16x16 a1 = load_afrag(arow1 + k);
#pragma unroll
        for (int j = 0; j < 4; ++j) {
            const __bf16* bp = Wt + (size_t)(n0 + 16 * j + lm) * D_ + k + hf * 16;
            bf16x16 bfr = load_bfrag(bp);
            acc[0][j] = wmma_bf16(a0, bfr, acc[0][j]);
            acc[1][j] = wmma_bf16(a1, bfr, acc[1][j]);
        }
    }
#pragma unroll
    for (int j = 0; j < 4; ++j) {
        int n = n0 + 16 * j + lm;
        float bv = bias[n];
#pragma unroll
        for (int t = 0; t < 2; ++t)
#pragma unroll
            for (int r = 0; r < 8; ++r) {
                int row = m0 + 16 * t + r + 8 * hf;
                C[(size_t)row * D_ + n] = f2bf(acc[t][j][r] + bv);
            }
    }
}

// ---------------------------------------------------------------------------
// 4) Per-head V transpose: vt[(bh*64+d)*S + s] = v[(b*S+s)*D + h*64 + d]
//    block (32,8), grid (64, 2, 32)
// ---------------------------------------------------------------------------
__global__ void vtrans(const __bf16* __restrict__ v, __bf16* __restrict__ vt) {
    int bh = blockIdx.z;
    int b = bh >> 4, h = bh & 15;
    __shared__ __bf16 tile[32][33];
    int s0 = blockIdx.x * 32, d0 = blockIdx.y * 32;
    int tx = threadIdx.x, ty = threadIdx.y;
#pragma unroll
    for (int j = 0; j < 4; ++j)
        tile[ty + 8 * j][tx] =
            v[(size_t)(b * S_ + s0 + ty + 8 * j) * D_ + h * HD_ + d0 + tx];
    __syncthreads();
#pragma unroll
    for (int j = 0; j < 4; ++j)
        vt[(size_t)(bh * HD_ + d0 + ty + 8 * j) * S_ + s0 + tx] = tile[tx][ty + 8 * j];
}

// ---------------------------------------------------------------------------
// 5) Flash attention: per wave one 16-row Q tile of one (b,h); loop 32 keys.
//    grid (32, 32), block 128 (4 independent waves)
// ---------------------------------------------------------------------------
__global__ void attn(const __bf16* __restrict__ q,
                     const __bf16* __restrict__ k,
                     const __bf16* __restrict__ vt,
                     __bf16* __restrict__ ctx) {
    int bh = blockIdx.x;
    int b = bh >> 4, h = bh & 15;
    int wave = threadIdx.x >> 5;
    int lane = threadIdx.x & 31;
    int lm = lane & 15, hf = lane >> 4;
    int q0 = (blockIdx.y * 4 + wave) * 16;

    __shared__ __bf16 pbuf[4][16 * 32];
    __bf16* P = pbuf[wave];

    // Q fragments for d=0..31 and d=32..63 (hoisted out of the key loop)
    const __bf16* qrow = q + (size_t)(b * S_ + q0 + lm) * D_ + h * HD_ + hf * 16;
    bf16x16 qa0 = load_afrag(qrow);
    bf16x16 qa1 = load_afrag(qrow + 32);

    float m[8], l[8];
#pragma unroll
    for (int r = 0; r < 8; ++r) { m[r] = -1e30f; l[r] = 0.f; }
    f32x8 acc[4] = {};

    const float scale = 0.125f * 1.44269504f;   // 1/sqrt(64) * log2(e)

    for (int s0 = 0; s0 < S_; s0 += 32) {
        // ---- scores: two 16x16 tiles (keys s0..+15, s0+16..+31) ----
        f32x8 sc0 = {}, sc1 = {};
        {
            const __bf16* kb0 = k + (size_t)(b * S_ + s0 + lm) * D_ + h * HD_ + hf * 16;
            sc0 = wmma_bf16(qa0, load_bfrag(kb0), sc0);
            sc0 = wmma_bf16(qa1, load_bfrag(kb0 + 32), sc0);
            const __bf16* kb1 = k + (size_t)(b * S_ + s0 + 16 + lm) * D_ + h * HD_ + hf * 16;
            sc1 = wmma_bf16(qa0, load_bfrag(kb1), sc1);
            sc1 = wmma_bf16(qa1, load_bfrag(kb1 + 32), sc1);
        }
        // ---- online softmax (base-2); rows r+8*hf live in this lane ----
        float alpha[8];
#pragma unroll
        for (int r = 0; r < 8; ++r) {
            float a0 = sc0[r] * scale;
            float a1 = sc1[r] * scale;
            float mx = fmaxf(a0, a1);
#pragma unroll
            for (int off = 1; off < 16; off <<= 1)
                mx = fmaxf(mx, __shfl_xor(mx, off, 32));
            float mnew = fmaxf(m[r], mx);
            float al = exp2f(m[r] - mnew);
            float p0 = exp2f(a0 - mnew);
            float p1 = exp2f(a1 - mnew);
            float rs = p0 + p1;
#pragma unroll
            for (int off = 1; off < 16; off <<= 1)
                rs += __shfl_xor(rs, off, 32);
            l[r] = l[r] * al + rs;
            m[r] = mnew;
            alpha[r] = al;
            int row = r + 8 * hf;
            P[row * 32 + lm]      = f2bf(p0);
            P[row * 32 + 16 + lm] = f2bf(p1);
        }
        __syncthreads();
        // rescale running context
#pragma unroll
        for (int j = 0; j < 4; ++j)
#pragma unroll
            for (int r = 0; r < 8; ++r) acc[j][r] *= alpha[r];
        // P as A-fragment (16x32) from LDS
        bf16x16 pa = load_afrag(P + lm * 32 + hf * 16);
        // ---- ctx += P @ V   (V^T rows are contiguous in keys) ----
#pragma unroll
        for (int j = 0; j < 4; ++j) {
            const __bf16* vb =
                vt + (size_t)(bh * HD_ + 16 * j + lm) * S_ + s0 + hf * 16;
            acc[j] = wmma_bf16(pa, load_bfrag(vb), acc[j]);
        }
        __syncthreads();
    }
    // ---- normalize and store ctx (bf16) ----
    float linv[8];
#pragma unroll
    for (int r = 0; r < 8; ++r) linv[r] = 1.0f / l[r];
#pragma unroll
    for (int j = 0; j < 4; ++j) {
        int n = h * HD_ + 16 * j + lm;
#pragma unroll
        for (int r = 0; r < 8; ++r) {
            int row = q0 + r + 8 * hf;
            ctx[(size_t)(b * S_ + row) * D_ + n] = f2bf(acc[j][r] * linv[r]);
        }
    }
}

// ---------------------------------------------------------------------------
// 6) Output projection + residual: out = x + ctx @ Wo + bo   (f32 out)
//    wave computes 32x64; grid (128, 4)
// ---------------------------------------------------------------------------
__global__ void gemm_out(const __bf16* __restrict__ A,
                         const __bf16* __restrict__ Wt,
                         const float* __restrict__ bias,
                         const float* __restrict__ xres,
                         float* __restrict__ out) {
    int m0   = blockIdx.x * 32;
    int wave = threadIdx.x >> 5;
    int lane = threadIdx.x & 31;
    int lm = lane & 15, hf = lane >> 4;
    int n0 = blockIdx.y * 256 + wave * 64;

    const __bf16* arow0 = A + (size_t)(m0 + lm) * D_ + hf * 16;
    const __bf16* arow1 = arow0 + (size_t)16 * D_;
    f32x8 acc[2][4] = {};
    for (int kk = 0; kk < D_; kk += 32) {
        bf16x16 a0 = load_afrag(arow0 + kk);
        bf16x16 a1 = load_afrag(arow1 + kk);
#pragma unroll
        for (int j = 0; j < 4; ++j) {
            const __bf16* bp = Wt + (size_t)(n0 + 16 * j + lm) * D_ + kk + hf * 16;
            bf16x16 bfr = load_bfrag(bp);
            acc[0][j] = wmma_bf16(a0, bfr, acc[0][j]);
            acc[1][j] = wmma_bf16(a1, bfr, acc[1][j]);
        }
    }
#pragma unroll
    for (int j = 0; j < 4; ++j) {
        int n = n0 + 16 * j + lm;
        float bv = bias[n];
#pragma unroll
        for (int t = 0; t < 2; ++t)
#pragma unroll
            for (int r = 0; r < 8; ++r) {
                int row = m0 + 16 * t + r + 8 * hf;
                out[(size_t)row * D_ + n] = xres[(size_t)row * D_ + n] + acc[t][j][r] + bv;
            }
    }
}

// ---------------------------------------------------------------------------
// Launch
// ---------------------------------------------------------------------------
extern "C" void kernel_launch(void* const* d_in, const int* in_sizes, int n_in,
                              void* d_out, int out_size, void* d_ws, size_t ws_size,
                              hipStream_t stream) {
    const float* x  = (const float*)d_in[0];
    const float* Wq = (const float*)d_in[1];
    const float* bq = (const float*)d_in[2];
    const float* Wk = (const float*)d_in[3];
    const float* bk = (const float*)d_in[4];
    const float* Wv = (const float*)d_in[5];
    const float* bv = (const float*)d_in[6];
    const float* Wo = (const float*)d_in[7];
    const float* bo = (const float*)d_in[8];
    const float* g  = (const float*)d_in[9];
    const float* be = (const float*)d_in[10];

    char* ws = (char*)d_ws;
    const size_t MB = 1024 * 1024;
    __bf16* xn  = (__bf16*)(ws + 0 * MB);
    __bf16* qb  = (__bf16*)(ws + 8 * MB);
    __bf16* kb  = (__bf16*)(ws + 16 * MB);
    __bf16* vb  = (__bf16*)(ws + 24 * MB);
    __bf16* vtb = (__bf16*)(ws + 32 * MB);
    __bf16* cb  = (__bf16*)(ws + 40 * MB);
    __bf16* WqT = (__bf16*)(ws + 48 * MB);
    __bf16* WkT = (__bf16*)(ws + 50 * MB);
    __bf16* WvT = (__bf16*)(ws + 52 * MB);
    __bf16* WoT = (__bf16*)(ws + 54 * MB);

    ln_kernel<<<B_ * S_, 256, 0, stream>>>(x, g, be, xn);

    dim3 tb(32, 8);
    wtrans<<<dim3(32, 32), tb, 0, stream>>>(Wq, WqT);
    wtrans<<<dim3(32, 32), tb, 0, stream>>>(Wk, WkT);
    wtrans<<<dim3(32, 32), tb, 0, stream>>>(Wv, WvT);
    wtrans<<<dim3(32, 32), tb, 0, stream>>>(Wo, WoT);

    dim3 gg(128, 4);
    gemm_qkv<<<gg, 128, 0, stream>>>(xn, WqT, bq, qb);
    gemm_qkv<<<gg, 128, 0, stream>>>(xn, WkT, bk, kb);
    gemm_qkv<<<gg, 128, 0, stream>>>(xn, WvT, bv, vb);

    vtrans<<<dim3(64, 2, 32), tb, 0, stream>>>(vb, vtb);

    attn<<<dim3(32, 32), 128, 0, stream>>>(qb, kb, vtb, cb);

    gemm_out<<<gg, 128, 0, stream>>>(cb, WoT, bo, x, (float*)d_out);
}
